// CHyperSoftmaxLayer_62182536511711
// MI455X (gfx1250) — matrix operand order repaired and checked by reading, not compiled, not run on hardware
//
#include <hip/hip_runtime.h>
#include <hip/hip_bf16.h>

typedef __attribute__((ext_vector_type(16))) _Float16 v16h;
typedef __attribute__((ext_vector_type(8)))  _Float16 v8h;
typedef __attribute__((ext_vector_type(8)))  float    v8f;

#define BB    8192
#define DD    2048
#define CC    1000
#define CP    1024
#define HH    16
#define LUT_N 2048

// workspace layout (bytes)
#define OFF_XH  0u
#define OFF_WH  (OFF_XH + (size_t)BB * DD * 2)          // 32 MB
#define OFF_SX  (OFF_WH + (size_t)CP * DD * 2)          // +4 MB
#define OFF_SW  (OFF_SX + (size_t)BB * 4)
#define OFF_LUT (OFF_SW + (size_t)CP * 4)

// ---------------------------------------------------------------------------
// Pass 1: convert a row-major f32 matrix to f16 and record per-row
// rsqrt(sum(x^2)) (the l2-norm scale). Rows >= validRows are zero-padded.
// ---------------------------------------------------------------------------
__global__ __launch_bounds__(256)
void rownorm_f16_kernel(const float* __restrict__ src, _Float16* __restrict__ dst,
                        float* __restrict__ scale, int validRows) {
    const int row = blockIdx.x;
    const int t   = threadIdx.x;            // 256 threads, DD/256 = 8 elems each
    float s = 0.f;
    _Float16* dp = dst + (size_t)row * DD;
    if (row < validRows) {
        const float* rp = src + (size_t)row * DD;
        #pragma unroll
        for (int i = 0; i < DD / 256; ++i) {
            float v = rp[t + i * 256];
            s += v * v;
            dp[t + i * 256] = (_Float16)v;
        }
    } else {
        #pragma unroll
        for (int i = 0; i < DD / 256; ++i) dp[t + i * 256] = (_Float16)0.f;
    }
    __shared__ float sb[256];
    sb[t] = s;
    __syncthreads();
    #pragma unroll
    for (int o = 128; o > 0; o >>= 1) {
        if (t < o) sb[t] += sb[t + o];
        __syncthreads();
    }
    if (t == 0)
        scale[row] = (row < validRows) ? rsqrtf(fmaxf(sb[0], 1e-12f)) : 0.f;
}

// ---------------------------------------------------------------------------
// Pass 2: tabulate the scalar piecewise-linear MLP g(s), s in [-1, 1].
// g(s) = relu( relu6( relu6(s*w1 + b1) @ w2 + b2 ) @ w3 + b3 )
// ---------------------------------------------------------------------------
__global__ __launch_bounds__(256)
void build_lut_kernel(const float* __restrict__ w1, const float* __restrict__ b1,
                      const float* __restrict__ w2, const float* __restrict__ b2,
                      const float* __restrict__ w3, const float* __restrict__ b3,
                      float* __restrict__ lut) {
    const int j = blockIdx.x * blockDim.x + threadIdx.x;
    if (j >= LUT_N) return;
    const float s = -1.f + 2.f * (float)j / (float)(LUT_N - 1);
    float h1[HH], h2[HH];
    #pragma unroll
    for (int k = 0; k < HH; ++k)
        h1[k] = fminf(fmaxf(s * w1[k] + b1[k], 0.f), 6.f);
    #pragma unroll
    for (int k = 0; k < HH; ++k) {
        float a = b2[k];
        #pragma unroll
        for (int h = 0; h < HH; ++h) a += h1[h] * w2[h * HH + k];
        h2[k] = fminf(fmaxf(a, 0.f), 6.f);
    }
    float o = b3[0];
    #pragma unroll
    for (int h = 0; h < HH; ++h) o += h2[h] * w3[h];
    lut[j] = fmaxf(o, 0.f);
}

// ---------------------------------------------------------------------------
// Pass 3: fused WMMA GEMM (16 rows x 1024 cols per workgroup, K=2048) +
// cosine scaling + LUT-MLP + row softmax. 16 waves/WG, 4 N-tiles per wave.
// ---------------------------------------------------------------------------
__global__ __launch_bounds__(512)
void fused_gemm_mlp_softmax_kernel(const _Float16* __restrict__ xh,
                                   const _Float16* __restrict__ wh,
                                   const float* __restrict__ sx,
                                   const float* __restrict__ sw,
                                   const float* __restrict__ lutg,
                                   float* __restrict__ out) {
    __shared__ float slut[LUT_N];
    __shared__ float red[16][17];
    __shared__ float rowmax[16];
    __shared__ float rowsum[16];

    const int tid = threadIdx.x;
    #pragma unroll
    for (int i = 0; i < LUT_N / 512; ++i) slut[tid + i * 512] = lutg[tid + i * 512];

    const int wave  = tid >> 5;
    const int lane  = tid & 31;
    const int half  = lane >> 4;     // hi/lo 16-lane group
    const int l15   = lane & 15;
    const int m0    = blockIdx.x * 16;
    const int nBase = wave * 64;

    // A fragment source: row (m0+l15), K chunks at {8*half, 16+8*half}
    const _Float16* aRow = xh + (size_t)(m0 + l15) * DD + half * 8;
    // B fragment sources: column n = row of W, K chunks at {8*half, 16+8*half}
    const _Float16* bCol[4];
    #pragma unroll
    for (int t = 0; t < 4; ++t)
        bCol[t] = wh + (size_t)(nBase + t * 16 + l15) * DD + half * 8;

    v8f acc[4];
    #pragma unroll
    for (int t = 0; t < 4; ++t) acc[t] = (v8f){};

    for (int k = 0; k < DD; k += 32) {
        v8h a0 = *(const v8h*)(aRow + k);
        v8h a1 = *(const v8h*)(aRow + k + 16);
        v16h A = __builtin_shufflevector(a0, a1, 0,1,2,3,4,5,6,7,8,9,10,11,12,13,14,15);
        #pragma unroll
        for (int t = 0; t < 4; ++t) {
            v8h b0 = *(const v8h*)(bCol[t] + k);
            v8h b1 = *(const v8h*)(bCol[t] + k + 16);
            v16h Bm = __builtin_shufflevector(b0, b1, 0,1,2,3,4,5,6,7,8,9,10,11,12,13,14,15);
            acc[t] = __builtin_amdgcn_wmma_f32_16x16x32_f16(
                false, A, false, Bm, (short)0, acc[t], false, false);
        }
    }
    __syncthreads();   // slut populated; acc complete

    // ---- cosine scale + LUT lerp; track per-row max (8 rows per lane) ----
    const int rbase = half ? 8 : 0;
    float sxr[8];
    #pragma unroll
    for (int i = 0; i < 8; ++i) sxr[i] = sx[m0 + rbase + i];

    float mymax[8];
    #pragma unroll
    for (int i = 0; i < 8; ++i) mymax[i] = -1e30f;

    #pragma unroll
    for (int t = 0; t < 4; ++t) {
        const int c = nBase + t * 16 + l15;
        const float scw = sw[c];
        #pragma unroll
        for (int i = 0; i < 8; ++i) {
            float s = acc[t][i] * sxr[i] * scw;          // cosine similarity
            s = fminf(fmaxf(s, -1.f), 1.f);
            float tt = (s + 1.f) * (0.5f * (float)(LUT_N - 1));
            int ix = (int)tt;
            ix = ix > (LUT_N - 2) ? (LUT_N - 2) : ix;
            float fr = tt - (float)ix;
            float g = slut[ix] + fr * (slut[ix + 1] - slut[ix]);
            if (c >= CC) g = -1e30f;                     // mask pad columns
            acc[t][i] = g;
            mymax[i] = fmaxf(mymax[i], g);
        }
    }
    // reduce max over the 16-lane half (covers all 64 cols of this wave)
    #pragma unroll
    for (int i = 0; i < 8; ++i)
        #pragma unroll
        for (int o = 1; o < 16; o <<= 1)
            mymax[i] = fmaxf(mymax[i], __shfl_xor(mymax[i], o, 32));
    if (l15 == 0) {
        #pragma unroll
        for (int i = 0; i < 8; ++i) red[rbase + i][wave] = mymax[i];
    }
    __syncthreads();
    if (tid < 16) {
        float m = red[tid][0];
        #pragma unroll
        for (int w = 1; w < 16; ++w) m = fmaxf(m, red[tid][w]);
        rowmax[tid] = m;
    }
    __syncthreads();

    // ---- exp + row sums ----
    float mysum[8];
    #pragma unroll
    for (int i = 0; i < 8; ++i) mysum[i] = 0.f;
    #pragma unroll
    for (int t = 0; t < 4; ++t)
        #pragma unroll
        for (int i = 0; i < 8; ++i) {
            float e = __expf(acc[t][i] - rowmax[rbase + i]);  // masked -> 0
            acc[t][i] = e;
            mysum[i] += e;
        }
    #pragma unroll
    for (int i = 0; i < 8; ++i)
        #pragma unroll
        for (int o = 1; o < 16; o <<= 1)
            mysum[i] += __shfl_xor(mysum[i], o, 32);
    if (l15 == 0) {
        #pragma unroll
        for (int i = 0; i < 8; ++i) red[rbase + i][wave] = mysum[i];
    }
    __syncthreads();
    if (tid < 16) {
        float s = red[tid][0];
        #pragma unroll
        for (int w = 1; w < 16; ++w) s += red[tid][w];
        rowsum[tid] = s;
    }
    __syncthreads();

    // ---- normalize & store (only real classes) ----
    float inv[8];
    #pragma unroll
    for (int i = 0; i < 8; ++i) inv[i] = 1.f / rowsum[rbase + i];
    #pragma unroll
    for (int t = 0; t < 4; ++t) {
        const int c = nBase + t * 16 + l15;
        if (c < CC) {
            #pragma unroll
            for (int i = 0; i < 8; ++i)
                out[(size_t)(m0 + rbase + i) * CC + c] = acc[t][i] * inv[i];
        }
    }
}

// ---------------------------------------------------------------------------
extern "C" void kernel_launch(void* const* d_in, const int* in_sizes, int n_in,
                              void* d_out, int out_size, void* d_ws, size_t ws_size,
                              hipStream_t stream) {
    (void)in_sizes; (void)n_in; (void)out_size; (void)ws_size;
    const float* x  = (const float*)d_in[0];
    const float* W  = (const float*)d_in[1];
    const float* w1 = (const float*)d_in[2];
    const float* b1 = (const float*)d_in[3];
    const float* w2 = (const float*)d_in[4];
    const float* b2 = (const float*)d_in[5];
    const float* w3 = (const float*)d_in[6];
    const float* b3 = (const float*)d_in[7];
    float* out = (float*)d_out;

    char* ws = (char*)d_ws;
    _Float16* xh  = (_Float16*)(ws + OFF_XH);
    _Float16* wh  = (_Float16*)(ws + OFF_WH);
    float*    sx  = (float*)(ws + OFF_SX);
    float*    sw  = (float*)(ws + OFF_SW);
    float*    lut = (float*)(ws + OFF_LUT);

    rownorm_f16_kernel<<<BB, 256, 0, stream>>>(x, xh, sx, BB);
    rownorm_f16_kernel<<<CP, 256, 0, stream>>>(W, wh, sw, CC);
    build_lut_kernel<<<LUT_N / 256, 256, 0, stream>>>(w1, b1, w2, b2, w3, b3, lut);
    fused_gemm_mlp_softmax_kernel<<<BB / 16, 512, 0, stream>>>(xh, wh, sx, sw, lut, out);
}
